// DefenseTrajectoryPredictor_73186242723996
// MI455X (gfx1250) — matrix operand order repaired and checked by reading, not compiled, not run on hardware
//
#include <hip/hip_runtime.h>
#include <cstdint>
#include <cstddef>

#define B_SZ    1024
#define T_SEQ   250
#define IN_DIM  158
#define IN_PAD  160
#define HID     128
#define GATES   512
#define PROJ    64
#define OUT_D   22

typedef __attribute__((ext_vector_type(16))) __bf16 v16bf;
typedef __attribute__((ext_vector_type(8)))  float  v8f;

union BFrag { uint4 q[2]; unsigned short s[16]; v16bf v; };

__device__ __forceinline__ unsigned short f2bf(float f) {
  unsigned int u = __float_as_uint(f);
  u += 0x7FFFu + ((u >> 16) & 1u);          // round-to-nearest-even
  return (unsigned short)(u >> 16);
}

__device__ __forceinline__ float fsig(float x) { return 1.0f / (1.0f + __expf(-x)); }
__device__ __forceinline__ float ftanh_(float x) {
  float e = __expf(2.0f * x);
  return 1.0f - 2.0f / (e + 1.0f);          // stable: ->1 / ->-1 at +/-inf
}

// A-fragment (16x32 bf16): lane<16 holds K[0..7],K[16..23]; lane>=16 holds K[8..15],K[24..31]
__device__ __forceinline__ v16bf ld_fragA(const unsigned short* rowbase, int kt, int hlf) {
  BFrag u;
  const unsigned short* p = rowbase + kt * 32 + hlf * 8;
  u.q[0] = *(const uint4*)(p);
  u.q[1] = *(const uint4*)(p + 16);
  return u.v;
}
// B-fragment (32x16 bf16): lane holds column (lane&15); lane<16 K[0..7],K[16..23]; lane>=16 K[8..15],K[24..31]
__device__ __forceinline__ v16bf ld_fragB(const unsigned short* nrowbase, int kt, int hlf) {
  BFrag u;
  const unsigned short* p = nrowbase + kt * 32 + hlf * 8;
  u.q[0] = *(const uint4*)(p);
  u.q[1] = *(const uint4*)(p + 16);
  return u.v;
}

#define WMMA_BF16(A, Bv, C) \
  __builtin_amdgcn_wmma_f32_16x16x32_bf16(false, (A), false, (Bv), (short)0, (C), false, false)

// ---------------------------------------------------------------- kernel 0: LN stats
__global__ __launch_bounds__(256) void ln_stats_kernel(const float* __restrict__ x,
                                                       float* __restrict__ stats) {
  const int nrows = B_SZ * T_SEQ;
  const int row = blockIdx.x * 8 + (threadIdx.x >> 5);
  const int lane = threadIdx.x & 31;
  if (row >= nrows) return;
  const float* rp = x + (size_t)row * IN_DIM;
  float s = 0.f, s2 = 0.f;
  for (int k = lane; k < IN_DIM; k += 32) { float v = rp[k]; s += v; s2 += v * v; }
#pragma unroll
  for (int off = 16; off > 0; off >>= 1) {
    s  += __shfl_xor(s,  off, 32);
    s2 += __shfl_xor(s2, off, 32);
  }
  if (lane == 0) {
    float mu  = s * (1.0f / IN_DIM);
    float var = s2 * (1.0f / IN_DIM) - mu * mu;
    stats[(size_t)row * 2 + 0] = mu;
    stats[(size_t)row * 2 + 1] = rsqrtf(var + 1e-5f);
  }
}

// ------------------------------------------------- kernels 1/2: fused (LN+)LSTM layer
// One workgroup (8 waves) per 16-row batch tile; weights live in LDS for the whole
// T-loop (only 1 WG fits per WGP -> exactly 2 waves/SIMD, so launch_bounds(256,2)
// maximizes the VGPR budget). Wave w owns hidden cols [16w,16w+16) of all 4 gates ->
// in-register cell update. Recurrent W_hh B-fragments (16 frags = 128 VGPRs) stay in
// registers for the whole sequence -> the 16 recurrent WMMAs per step are register-only.
template <bool FIRST, int KINPAD>
__global__ __launch_bounds__(256, 2) void lstm_layer_kernel(
    const float* __restrict__ x_f32,            // FIRST: raw x [B][T][158]
    const unsigned short* __restrict__ x_bf16,  // !FIRST: h_prev_seq bf16 [B][T][128]
    const float* __restrict__ stats,            // FIRST: [B*T][2] mu,rstd
    const float* __restrict__ ln_gamma,
    const float* __restrict__ ln_beta,
    const float* __restrict__ W_ih,             // [512][KIN] f32
    const float* __restrict__ W_hh,             // [512][128] f32
    const float* __restrict__ bias,             // [512] f32
    unsigned short* __restrict__ h_seq_out) {   // bf16 [B][T][128]
  constexpr int KIN   = FIRST ? IN_DIM : HID;
  constexpr int KT_IN = KINPAD / 32;

  extern __shared__ unsigned char smem_raw[];
  unsigned short* sWih = (unsigned short*)smem_raw;   // [GATES][KINPAD] bf16
  unsigned short* sWhh = sWih + GATES * KINPAD;       // [GATES][HID]    bf16
  unsigned short* sH   = sWhh + GATES * HID;          // [2][16][HID]    bf16 ping-pong
  float* sGamma = (float*)(sH + 2 * 16 * HID);        // [KINPAD] (FIRST only)
  float* sBeta  = sGamma + KINPAD;

  const int tid = threadIdx.x;
  for (int i = tid; i < GATES * KINPAD; i += 256) {
    int r = i / KINPAD, c = i - r * KINPAD;
    sWih[i] = (c < KIN) ? f2bf(W_ih[(size_t)r * KIN + c]) : (unsigned short)0;
  }
  for (int i = tid; i < GATES * HID; i += 256) sWhh[i] = f2bf(W_hh[i]);
  for (int i = tid; i < 2 * 16 * HID; i += 256) sH[i] = 0;
  if constexpr (FIRST) {
    for (int i = tid; i < KINPAD; i += 256) {
      sGamma[i] = (i < KIN) ? ln_gamma[i] : 0.0f;
      sBeta[i]  = (i < KIN) ? ln_beta[i]  : 0.0f;
    }
  }
  __syncthreads();

  const int wave  = tid >> 5;
  const int lane  = tid & 31;
  const int hlf   = lane >> 4;
  const int l16   = lane & 15;
  const int btile = blockIdx.x;
  const int brow  = btile * 16 + l16;   // batch row this lane feeds as A operand

  float bg[4];
#pragma unroll
  for (int g = 0; g < 4; ++g) bg[g] = bias[g * HID + wave * 16 + l16];

  // hoist all recurrent B-fragments into registers for the whole sequence
  v16bf bhh[4][4];
#pragma unroll
  for (int kt = 0; kt < 4; ++kt)
#pragma unroll
    for (int g = 0; g < 4; ++g)
      bhh[kt][g] = ld_fragB(sWhh + (size_t)(g * HID + wave * 16 + l16) * HID, kt, hlf);

  v8f cst = {};   // persistent cell state: 8 cells (rows r+8*hlf, col wave*16+l16)
  int p = 0;
  for (int t = 0; t < T_SEQ; ++t) {
    const unsigned short* hRow = sH + p * (16 * HID) + l16 * HID;

    // 1) prefetch/build input-A fragments (global loads overlap recurrent WMMAs below)
    v16bf ax[KT_IN];
    if constexpr (FIRST) {
      const size_t rb = ((size_t)brow * T_SEQ + t) * IN_DIM;
      const size_t st = ((size_t)brow * T_SEQ + t) * 2;
      const float mu = stats[st + 0];
      const float rs = stats[st + 1];
#pragma unroll
      for (int kt = 0; kt < KT_IN; ++kt) {
        BFrag u;
        const bool full = (kt * 32 + 31) < KIN;   // compile-time per unrolled kt
#pragma unroll
        for (int e = 0; e < 16; ++e) {
          const int k = kt * 32 + ((e < 8) ? (hlf * 8 + e) : (16 + hlf * 8 + e - 8));
          float v;
          if (full) {
            v = (x_f32[rb + k] - mu) * rs * sGamma[k] + sBeta[k];
          } else {
            v = (k < KIN) ? ((x_f32[rb + k] - mu) * rs * sGamma[k] + sBeta[k]) : 0.0f;
          }
          u.s[e] = f2bf(v);
        }
        ax[kt] = u.v;
      }
    } else {
      const unsigned short* rp = x_bf16 + ((size_t)brow * T_SEQ + t) * HID;
#pragma unroll
      for (int kt = 0; kt < KT_IN; ++kt) ax[kt] = ld_fragA(rp, kt, hlf);
    }

    v8f acc[4];
#pragma unroll
    for (int g = 0; g < 4; ++g) { v8f z = {}; acc[g] = z; }

    // 2) recurrent half: A per k-tile from LDS, B resident in registers
#pragma unroll
    for (int kt = 0; kt < 4; ++kt) {
      v16bf a = ld_fragA(hRow, kt, hlf);
      acc[0] = WMMA_BF16(a, bhh[kt][0], acc[0]);
      acc[1] = WMMA_BF16(a, bhh[kt][1], acc[1]);
      acc[2] = WMMA_BF16(a, bhh[kt][2], acc[2]);
      acc[3] = WMMA_BF16(a, bhh[kt][3], acc[3]);
    }

    // 3) input projection: per k-tile, group all 4 B-frag loads before 4 WMMAs
#pragma unroll
    for (int kt = 0; kt < KT_IN; ++kt) {
      v16bf b0 = ld_fragB(sWih + (size_t)(0 * HID + wave * 16 + l16) * KINPAD, kt, hlf);
      v16bf b1 = ld_fragB(sWih + (size_t)(1 * HID + wave * 16 + l16) * KINPAD, kt, hlf);
      v16bf b2 = ld_fragB(sWih + (size_t)(2 * HID + wave * 16 + l16) * KINPAD, kt, hlf);
      v16bf b3 = ld_fragB(sWih + (size_t)(3 * HID + wave * 16 + l16) * KINPAD, kt, hlf);
      acc[0] = WMMA_BF16(ax[kt], b0, acc[0]);
      acc[1] = WMMA_BF16(ax[kt], b1, acc[1]);
      acc[2] = WMMA_BF16(ax[kt], b2, acc[2]);
      acc[3] = WMMA_BF16(ax[kt], b3, acc[3]);
    }

    // 4) in-register LSTM cell (f32), write h (bf16) into other ping-pong buffer
    unsigned short* hN = sH + (1 - p) * (16 * HID);
#pragma unroll
    for (int r = 0; r < 8; ++r) {
      float iv = fsig(acc[0][r] + bg[0]);
      float fv = fsig(acc[1][r] + bg[1]);
      float gv = ftanh_(acc[2][r] + bg[2]);
      float ov = fsig(acc[3][r] + bg[3]);
      float c  = fv * cst[r] + iv * gv;
      cst[r]   = c;
      float h  = ov * ftanh_(c);
      hN[(r + 8 * hlf) * HID + wave * 16 + l16] = f2bf(h);
    }
    __syncthreads();

    // 5) coalesced h_t -> global (16 rows x 128 bf16 = 4KB, one b128 per thread)
    {
      const int rr = tid >> 4;
      const int cc = tid & 15;
      uint4 v = *(const uint4*)(hN + rr * HID + cc * 8);
      *(uint4*)(h_seq_out + (((size_t)(btile * 16 + rr)) * T_SEQ + t) * HID + cc * 8) = v;
    }
    p ^= 1;
  }
}

// ---------------------------------------------------------------- kernel 3: MLP head
__global__ __launch_bounds__(128) void mlp_proj_kernel(
    const unsigned short* __restrict__ h_seq,   // bf16 [B*T][128]
    const float* __restrict__ Wp1, const float* __restrict__ bp1,
    const float* __restrict__ Wp2, const float* __restrict__ bp2,
    float* __restrict__ out) {                  // f32 [B*T][22]
  __shared__ __align__(16) unsigned short sW1[PROJ * HID];  // [64][128]
  __shared__ __align__(16) unsigned short sW2[32 * PROJ];   // [32][64], rows>=22 zero
  __shared__ __align__(16) unsigned short sP[4][16 * PROJ]; // per-wave relu staging

  const int tid = threadIdx.x;
  for (int i = tid; i < PROJ * HID; i += 128) sW1[i] = f2bf(Wp1[i]);
  for (int i = tid; i < 32 * PROJ; i += 128) {
    int r = i >> 6;
    sW2[i] = (r < OUT_D) ? f2bf(Wp2[i]) : (unsigned short)0;
  }
  __syncthreads();

  const int wave = tid >> 5, lane = tid & 31, hlf = lane >> 4, l16 = lane & 15;
  const size_t mtile = (size_t)blockIdx.x * 4 + wave;
  const size_t mrow  = mtile * 16 + l16;

  // GEMM1: [16,128] x [128,64]
  v16bf a1[4];
  const unsigned short* rp = h_seq + mrow * HID;
#pragma unroll
  for (int kt = 0; kt < 4; ++kt) a1[kt] = ld_fragA(rp, kt, hlf);

  v8f acc[4];
#pragma unroll
  for (int j = 0; j < 4; ++j) { v8f z = {}; acc[j] = z; }
#pragma unroll
  for (int kt = 0; kt < 4; ++kt) {
    v16bf b0 = ld_fragB(sW1 + (size_t)(0 * 16 + l16) * HID, kt, hlf);
    v16bf b1 = ld_fragB(sW1 + (size_t)(1 * 16 + l16) * HID, kt, hlf);
    v16bf b2 = ld_fragB(sW1 + (size_t)(2 * 16 + l16) * HID, kt, hlf);
    v16bf b3 = ld_fragB(sW1 + (size_t)(3 * 16 + l16) * HID, kt, hlf);
    acc[0] = WMMA_BF16(a1[kt], b0, acc[0]);
    acc[1] = WMMA_BF16(a1[kt], b1, acc[1]);
    acc[2] = WMMA_BF16(a1[kt], b2, acc[2]);
    acc[3] = WMMA_BF16(a1[kt], b3, acc[3]);
  }

  // bias + relu -> bf16 staging (for A-layout of GEMM2)
  unsigned short* pst = sP[wave];
#pragma unroll
  for (int j = 0; j < 4; ++j) {
    float bb = bp1[j * 16 + l16];
#pragma unroll
    for (int r = 0; r < 8; ++r) {
      float v = acc[j][r] + bb;
      v = v > 0.0f ? v : 0.0f;
      pst[(r + 8 * hlf) * PROJ + j * 16 + l16] = f2bf(v);
    }
  }
  __syncthreads();

  // GEMM2: [16,64] x [64,32(pad of 22)]
  v16bf a2[2];
#pragma unroll
  for (int kt = 0; kt < 2; ++kt) a2[kt] = ld_fragA(pst + l16 * PROJ, kt, hlf);
  v8f o2[2];
#pragma unroll
  for (int j = 0; j < 2; ++j) { v8f z = {}; o2[j] = z; }
#pragma unroll
  for (int kt = 0; kt < 2; ++kt) {
    v16bf b0 = ld_fragB(sW2 + (size_t)(0 * 16 + l16) * PROJ, kt, hlf);
    v16bf b1 = ld_fragB(sW2 + (size_t)(1 * 16 + l16) * PROJ, kt, hlf);
    o2[0] = WMMA_BF16(a2[kt], b0, o2[0]);
    o2[1] = WMMA_BF16(a2[kt], b1, o2[1]);
  }

#pragma unroll
  for (int j = 0; j < 2; ++j) {
    const int col = j * 16 + l16;
    if (col < OUT_D) {
      const float bb = bp2[col];
#pragma unroll
      for (int r = 0; r < 8; ++r) {
        const size_t m = mtile * 16 + (r + 8 * hlf);
        out[m * OUT_D + col] = o2[j][r] + bb;
      }
    }
  }
}

// --------------------------------------------------------------------------- launch
extern "C" void kernel_launch(void* const* d_in, const int* in_sizes, int n_in,
                              void* d_out, int out_size, void* d_ws, size_t ws_size,
                              hipStream_t stream) {
  (void)in_sizes; (void)n_in; (void)out_size; (void)ws_size;
  const float* x        = (const float*)d_in[0];
  const float* ln_gamma = (const float*)d_in[1];
  const float* ln_beta  = (const float*)d_in[2];
  const float* W_ih0    = (const float*)d_in[3];
  const float* W_hh0    = (const float*)d_in[4];
  const float* b0       = (const float*)d_in[5];
  const float* W_ih1    = (const float*)d_in[6];
  const float* W_hh1    = (const float*)d_in[7];
  const float* b1       = (const float*)d_in[8];
  const float* Wp1      = (const float*)d_in[9];
  const float* bp1      = (const float*)d_in[10];
  const float* Wp2      = (const float*)d_in[11];
  const float* bp2      = (const float*)d_in[12];
  float* out = (float*)d_out;

  char* ws = (char*)d_ws;
  const size_t rows = (size_t)B_SZ * T_SEQ;                 // 256000
  float* stats       = (float*)ws;                          // 2.05 MB
  unsigned short* h0 = (unsigned short*)(ws + rows * 2 * sizeof(float)); // 65.5 MB
  unsigned short* h1 = h0 + rows * HID;                     // 65.5 MB

  constexpr size_t SMEM_L0 =
      (size_t)(GATES * IN_PAD + GATES * HID + 2 * 16 * HID) * 2 + (size_t)IN_PAD * 2 * 4;
  constexpr size_t SMEM_L1 = (size_t)(GATES * HID + GATES * HID + 2 * 16 * HID) * 2;

  (void)hipFuncSetAttribute(reinterpret_cast<const void*>(&lstm_layer_kernel<true, IN_PAD>),
                            hipFuncAttributeMaxDynamicSharedMemorySize, (int)SMEM_L0);
  (void)hipFuncSetAttribute(reinterpret_cast<const void*>(&lstm_layer_kernel<false, HID>),
                            hipFuncAttributeMaxDynamicSharedMemorySize, (int)SMEM_L1);

  ln_stats_kernel<<<dim3((unsigned)((rows + 7) / 8)), dim3(256), 0, stream>>>(x, stats);

  lstm_layer_kernel<true, IN_PAD><<<dim3(B_SZ / 16), dim3(256), SMEM_L0, stream>>>(
      x, nullptr, stats, ln_gamma, ln_beta, W_ih0, W_hh0, b0, h0);

  lstm_layer_kernel<false, HID><<<dim3(B_SZ / 16), dim3(256), SMEM_L1, stream>>>(
      nullptr, h0, nullptr, nullptr, nullptr, W_ih1, W_hh1, b1, h1);

  mlp_proj_kernel<<<dim3((unsigned)(rows / 16 / 4)), dim3(128), 0, stream>>>(
      h1, Wp1, bp1, Wp2, bp2, out);
}
